// Encoder_12111807775221
// MI455X (gfx1250) — matrix operand order repaired and checked by reading, not compiled, not run on hardware
//
#include <hip/hip_runtime.h>
#include <hip/hip_bf16.h>
#include <math.h>

// ---------------- problem constants ----------------
#define BB 64
#define LL 256
#define DD 512
#define HH 512
#define KIN 1024           // 2*D (layer0 input) == 2*H (layer1 input)
#define G4H 2048           // 4*H
#define LABELV 1023

#define AS1 __attribute__((address_space(1)))
#define AS3 __attribute__((address_space(3)))

typedef _Float16 half8 __attribute__((ext_vector_type(8)));
typedef _Float16 v16h  __attribute__((ext_vector_type(16)));
typedef float    v8f   __attribute__((ext_vector_type(8)));
typedef int      v4i_g __attribute__((vector_size(16)));   // int4 (gcc-vector) for async builtin

#if __has_builtin(__builtin_amdgcn_global_load_async_to_lds_b128)
#define HAVE_ASYNC_LDS 1
#else
#define HAVE_ASYNC_LDS 0
#endif

// 16-byte global -> LDS copy: async DMA path on gfx1250 (ASYNCcnt), fallback plain.
__device__ __forceinline__ void copy16_g2l(const _Float16* g, _Float16* l) {
#if HAVE_ASYNC_LDS
  __builtin_amdgcn_global_load_async_to_lds_b128((AS1 v4i_g*)g, (AS3 v4i_g*)l, 0, 0);
#else
  *(half8*)l = *(const half8*)g;
#endif
}

__device__ __forceinline__ void async_join() {
#if HAVE_ASYNC_LDS
#if __has_builtin(__builtin_amdgcn_s_wait_asynccnt)
  __builtin_amdgcn_s_wait_asynccnt(0);
#else
  asm volatile("s_wait_asynccnt 0" ::: "memory");
#endif
#endif
  __syncthreads();
}

__device__ __forceinline__ half8 ld8(const _Float16* p) { return *(const half8*)p; }
__device__ __forceinline__ half8 ld8l(AS3 const _Float16* p) { return *(AS3 const half8*)p; }
__device__ __forceinline__ v16h mk16(half8 lo, half8 hi) {
  return __builtin_shufflevector(lo, hi, 0,1,2,3,4,5,6,7,8,9,10,11,12,13,14,15);
}
__device__ __forceinline__ float sigf(float x) { return 1.0f / (1.0f + expf(-x)); }

// ---- WMMA K-loop, A from global (row-major [64, lda] f16) ----
__device__ __forceinline__ void gemm_k(v8f* acc,
    const _Float16* __restrict__ A, int lda,
    const _Float16* __restrict__ W, int ldb,
    int n0, int K, int lc, int hi)
{
  const _Float16* brow = W + (size_t)(n0 + lc) * ldb;
  for (int kk = 0; kk < K; kk += 32) {
    v16h bf = mk16(ld8(brow + kk + hi * 8), ld8(brow + kk + hi * 8 + 16));
    __builtin_prefetch(brow + kk + 64, 0, 3);
#pragma unroll
    for (int i = 0; i < 4; ++i) {
      const _Float16* ap = A + (size_t)(i * 16 + lc) * lda + kk + hi * 8;
      v16h af = mk16(ld8(ap), ld8(ap + 16));
      acc[i] = __builtin_amdgcn_wmma_f32_16x16x32_f16(
          false, af, false, bf, (short)0, acc[i], false, false);
    }
  }
}

// ---- WMMA K-loop, A from LDS (ds_load_b128), B (weights) streamed from L2 ----
__device__ __forceinline__ void gemm_k_lds(v8f* acc,
    AS3 const _Float16* A, int lda,
    const _Float16* __restrict__ W, int ldb,
    int n0, int K, int lc, int hi)
{
  const _Float16* brow = W + (size_t)(n0 + lc) * ldb;
  for (int kk = 0; kk < K; kk += 32) {
    v16h bf = mk16(ld8(brow + kk + hi * 8), ld8(brow + kk + hi * 8 + 16));
    __builtin_prefetch(brow + kk + 64, 0, 3);
#pragma unroll
    for (int i = 0; i < 4; ++i) {
      AS3 const _Float16* ap = A + (i * 16 + lc) * lda + kk + hi * 8;
      v16h af = mk16(ld8l(ap), ld8l(ap + 16));
      acc[i] = __builtin_amdgcn_wmma_f32_16x16x32_f16(
          false, af, false, bf, (short)0, acc[i], false, false);
    }
  }
}

__device__ __forceinline__ void store_tiles(float* __restrict__ C, int ldc,
                                            int m0, int n0, const v8f* acc,
                                            int lc, int hi)
{
#pragma unroll
  for (int i = 0; i < 4; ++i)
#pragma unroll
    for (int v = 0; v < 8; ++v)
      C[(size_t)(m0 + i * 16 + hi * 8 + v) * ldc + n0 + lc] = acc[i][v];
}

// ---------------- small prep kernels ----------------

__global__ void cvt_f16_kernel(const float* __restrict__ src,
                               _Float16* __restrict__ dst, int count)
{
  int i = blockIdx.x * blockDim.x + threadIdx.x;
  if (i < count) dst[i] = (_Float16)src[i];
}

// Wt[n,k] = (f16) attn_W[k,n]   (512x512)
__global__ void transpose_w_kernel(const float* __restrict__ W,
                                   _Float16* __restrict__ Wt)
{
  int idx = blockIdx.x * blockDim.x + threadIdx.x;   // n*512 + k
  int n = idx >> 9, k = idx & 511;
  Wt[idx] = (_Float16)W[(k << 9) + n];
}

// gather word embeddings -> x0[l*B+b, 0:512] (f16), total[b,:] = sum_l emb
__global__ void embed_sum_kernel(const int* __restrict__ ids,
                                 const float* __restrict__ wemb,
                                 _Float16* __restrict__ x0,
                                 float* __restrict__ total)
{
  int b = blockIdx.x, tid = threadIdx.x;     // 64 blocks x 256 threads
  int d0 = tid, d1 = tid + 256;
  float a0 = 0.f, a1 = 0.f;
  for (int l = 0; l < LL; ++l) {
    int id = ids[b * LL + l];
    const float* w = wemb + (size_t)id * DD;
    float w0 = w[d0], w1 = w[d1];
    a0 += w0; a1 += w1;
    _Float16* xp = x0 + (size_t)(l * BB + b) * KIN;
    xp[d0] = (_Float16)w0; xp[d1] = (_Float16)w1;
  }
  total[b * DD + d0] = a0;
  total[b * DD + d1] = a1;
}

// tW[b,n] = sum_k total[b,k] * attn_W[k,n]
__global__ void tw_kernel(const float* __restrict__ total,
                          const float* __restrict__ W,
                          float* __restrict__ tW)
{
  int b = blockIdx.x;
  int n = blockIdx.y * blockDim.x + threadIdx.x;
  float acc = 0.f;
  const float* tb = total + b * DD;
  for (int k = 0; k < DD; ++k) acc += tb[k] * W[k * DD + n];
  tW[b * DD + n] = acc;
}

// qw[r,n] = sum_k x0[r,k] * Wt[n,k], r = l*B+b (M=16384, N=K=512)
// A slab (64 rows x 512 halfs) staged to LDS, shared by all 8 waves of the block.
__global__ void gemm_qw_kernel(const _Float16* __restrict__ x0,
                               const _Float16* __restrict__ Wt,
                               float* __restrict__ qw)
{
  __shared__ __align__(16) _Float16 Asl[64 * 512];   // 64 KB
  int wave = (blockIdx.x * blockDim.x + threadIdx.x) >> 5;  // 8192 waves
  int lane = threadIdx.x & 31, hi = lane >> 4, lc = lane & 15;
  int ntile = wave & 31;        // 32 n-tiles
  int mgrp  = wave >> 5;        // 256 m-slabs of 64 rows (== blockIdx.x >> 2)
  int n0 = ntile * 16, m0 = mgrp * 64;

  const _Float16* Ag = x0 + (size_t)m0 * KIN;
  for (int c = threadIdx.x; c < 64 * 64; c += blockDim.x) {   // 4096 16B chunks
    int row = c >> 6, ch = c & 63;
    copy16_g2l(Ag + (size_t)row * KIN + ch * 8, Asl + row * 512 + ch * 8);
  }
  async_join();

  v8f acc[4];
#pragma unroll
  for (int i = 0; i < 4; ++i) { v8f z; for (int e = 0; e < 8; ++e) z[e] = 0.f; acc[i] = z; }
  gemm_k_lds(acc, (AS3 const _Float16*)Asl, 512, Wt, DD, n0, DD, lc, hi);
  store_tiles(qw, DD, m0, n0, acc, lc, hi);
}

// 5-way factor attention -> x0[r, 512:1024]
__global__ void attn_kernel(const int* __restrict__ f1, const int* __restrict__ f2,
                            const int* __restrict__ f3, const int* __restrict__ f4,
                            const int* __restrict__ f5,
                            const float* __restrict__ femb,
                            const float* __restrict__ qw,
                            const float* __restrict__ tW,
                            _Float16* __restrict__ x0)
{
  int r = blockIdx.x;               // l*B + b
  int l = r >> 6, b = r & 63;
  int tid = threadIdx.x;            // 256
  int d0 = tid, d1 = tid + 256;
  int fid[5] = { f1[b * LL + l], f2[b * LL + l], f3[b * LL + l],
                 f4[b * LL + l], f5[b * LL + l] };
  const float inv = 1.0f / 255.0f;
  float q0 = (tW[b * DD + d0] - qw[(size_t)r * DD + d0]) * inv;
  float q1 = (tW[b * DD + d1] - qw[(size_t)r * DD + d1]) * inv;

  __shared__ float red[5][256];
#pragma unroll
  for (int k = 0; k < 5; ++k) {
    const float* fe = femb + (size_t)fid[k] * DD;
    red[k][tid] = q0 * fe[d0] + q1 * fe[d1];
  }
  __syncthreads();
  for (int off = 128; off > 0; off >>= 1) {
    if (tid < off)
#pragma unroll
      for (int k = 0; k < 5; ++k) red[k][tid] += red[k][tid + off];
    __syncthreads();
  }
  float sc[5], m = -3.4e38f;
#pragma unroll
  for (int k = 0; k < 5; ++k) {
    sc[k] = (fid[k] == LABELV) ? -1e9f : red[k][0];
    m = fmaxf(m, sc[k]);
  }
  float s = 0.f, p[5];
#pragma unroll
  for (int k = 0; k < 5; ++k) { p[k] = expf(sc[k] - m); s += p[k]; }
  float rs = 1.0f / s;
  float w0 = 0.f, w1 = 0.f;
#pragma unroll
  for (int k = 0; k < 5; ++k) {
    const float* fe = femb + (size_t)fid[k] * DD;
    float pk = p[k] * rs;
    w0 += pk * fe[d0]; w1 += pk * fe[d1];
  }
  _Float16* xp = x0 + (size_t)r * KIN + DD;
  xp[d0] = (_Float16)w0; xp[d1] = (_Float16)w1;
}

// ---------------- LSTM per-step kernels ----------------

// pre[dir][64][2048] = x_t @ Wih^T + h @ Whh^T + b (both dirs per launch).
// A slab (x_t: 128 KB, h: 64 KB) staged into LDS via async copies, shared by
// the block's 4 waves; weight rows stream from L2 (resident across steps).
__global__ void pregate_kernel(const _Float16* __restrict__ xin,   // [L*B, 1024]
                               const _Float16* __restrict__ hf16,  // [2, 64, 512]
                               const _Float16* __restrict__ wih,   // [2,2,2048,1024]
                               const _Float16* __restrict__ whh,   // [2,2,2048,512]
                               const float* __restrict__ bias,     // [2,2,2048]
                               float* __restrict__ pre,            // [2, 64, 2048]
                               int layer, int s)
{
  extern __shared__ __align__(16) _Float16 smem[];   // 64*1024 + 64*512 halfs
  _Float16* Ax = smem;                                // [64, 1024]
  _Float16* Ah = smem + 64 * KIN;                     // [64, 512]

  int wave = blockIdx.x * (blockDim.x >> 5) + (threadIdx.x >> 5); // 64 blks x 4 waves
  int lane = threadIdx.x & 31, hi = lane >> 4, lc = lane & 15;
  int dir = wave >> 7;            // uniform within a block (4 waves/block)
  int ntile = wave & 127;
  int n0 = ntile * 16;
  int t = dir ? (LL - 1 - s) : s;

  const _Float16* A0g = xin + (size_t)t * BB * KIN;           // contiguous 64*1024
  const _Float16* Ahg = hf16 + (size_t)dir * BB * HH;         // contiguous 64*512
  for (int c = threadIdx.x; c < (64 * KIN) / 8; c += blockDim.x)
    copy16_g2l(A0g + c * 8, Ax + c * 8);
  for (int c = threadIdx.x; c < (64 * HH) / 8; c += blockDim.x)
    copy16_g2l(Ahg + c * 8, Ah + c * 8);
  async_join();

  const _Float16* Wih = wih + (size_t)(layer * 2 + dir) * G4H * KIN;
  const _Float16* Whh = whh + (size_t)(layer * 2 + dir) * G4H * HH;
  float bv = bias[(layer * 2 + dir) * G4H + n0 + lc];

  v8f acc[4];
#pragma unroll
  for (int i = 0; i < 4; ++i) { v8f z; for (int e = 0; e < 8; ++e) z[e] = bv; acc[i] = z; }

  gemm_k_lds(acc, (AS3 const _Float16*)Ax, KIN, Wih, KIN, n0, KIN, lc, hi);
  gemm_k_lds(acc, (AS3 const _Float16*)Ah, HH,  Whh, HH,  n0, HH,  lc, hi);

  store_tiles(pre + (size_t)dir * BB * G4H, G4H, 0, n0, acc, lc, hi);
}

// elementwise gates: c' = sig(f)*c + sig(i)*tanh(g); h' = sig(o)*tanh(c')
__global__ void lstm_act_kernel(const float* __restrict__ pre,    // [2,64,2048]
                                float* __restrict__ cstate,       // [2,64,512]
                                _Float16* __restrict__ hf16,      // [2,64,512]
                                _Float16* __restrict__ yout_f16,  // layer0: y0, else null
                                float* __restrict__ yout_f32,     // layer1: d_out x, else null
                                float* __restrict__ hn_out,
                                float* __restrict__ cn_out,
                                int layer, int s)
{
  int idx = blockIdx.x * blockDim.x + threadIdx.x;  // 65536
  int j = idx & 511, b = (idx >> 9) & 63, dir = idx >> 15;
  int t = dir ? (LL - 1 - s) : s;
  const float* g = pre + (size_t)(dir * BB + b) * G4H;
  float gi = g[j], gf = g[HH + j], gg = g[2 * HH + j], go = g[3 * HH + j];
  int sidx = (dir * BB + b) * HH + j;
  float c = sigf(gf) * cstate[sidx] + sigf(gi) * tanhf(gg);
  float h = sigf(go) * tanhf(c);
  cstate[sidx] = c;
  hf16[sidx] = (_Float16)h;
  size_t ypos = (size_t)(t * BB + b) * KIN + dir * HH + j;
  if (yout_f16) yout_f16[ypos] = (_Float16)h;
  if (yout_f32) yout_f32[ypos] = h;
  if (s == LL - 1) {
    int o = (layer * 2 + dir) * BB * HH + b * HH + j;
    hn_out[o] = h;
    cn_out[o] = c;
  }
}

// ---------------- host orchestration ----------------
extern "C" void kernel_launch(void* const* d_in, const int* in_sizes, int n_in,
                              void* d_out, int out_size, void* d_ws, size_t ws_size,
                              hipStream_t stream) {
  (void)in_sizes; (void)n_in; (void)out_size; (void)ws_size;
  const int*   input_ids = (const int*)d_in[0];
  const int*   f1 = (const int*)d_in[1];
  const int*   f2 = (const int*)d_in[2];
  const int*   f3 = (const int*)d_in[3];
  const int*   f4 = (const int*)d_in[4];
  const int*   f5 = (const int*)d_in[5];
  const float* word_emb = (const float*)d_in[6];
  const float* feat_emb = (const float*)d_in[7];
  const float* attn_W   = (const float*)d_in[8];
  const float* w_ih     = (const float*)d_in[9];
  const float* w_hh     = (const float*)d_in[10];
  const float* bvec     = (const float*)d_in[11];
  float* out = (float*)d_out;   // [h_n(131072) | c_n(131072) | x(16777216)]

  // workspace carve (bytes)
  uint8_t* ws = (uint8_t*)d_ws;
  size_t off = 0;
  _Float16* x0   = (_Float16*)(ws + off); off += (size_t)LL * BB * KIN * 2;  // 32 MB
  _Float16* y0   = (_Float16*)(ws + off); off += (size_t)LL * BB * KIN * 2;  // 32 MB
  float*    qw   = (float*)   (ws + off); off += (size_t)LL * BB * DD * 4;   // 32 MB
  _Float16* wih16= (_Float16*)(ws + off); off += (size_t)2 * 2 * G4H * KIN * 2;
  _Float16* whh16= (_Float16*)(ws + off); off += (size_t)2 * 2 * G4H * HH * 2;
  _Float16* Wt   = (_Float16*)(ws + off); off += (size_t)DD * DD * 2;
  float*    total= (float*)   (ws + off); off += (size_t)BB * DD * 4;
  float*    tW   = (float*)   (ws + off); off += (size_t)BB * DD * 4;
  float*    pre  = (float*)   (ws + off); off += (size_t)2 * BB * G4H * 4;
  float*    cst  = (float*)   (ws + off); off += (size_t)2 * BB * HH * 4;
  _Float16* hf16 = (_Float16*)(ws + off); off += (size_t)2 * BB * HH * 2;

  cvt_f16_kernel<<<(2*2*G4H*KIN + 255)/256, 256, 0, stream>>>(w_ih, wih16, 2*2*G4H*KIN);
  cvt_f16_kernel<<<(2*2*G4H*HH  + 255)/256, 256, 0, stream>>>(w_hh, whh16, 2*2*G4H*HH);
  transpose_w_kernel<<<(DD*DD)/256, 256, 0, stream>>>(attn_W, Wt);

  embed_sum_kernel<<<BB, 256, 0, stream>>>(input_ids, word_emb, x0, total);
  tw_kernel<<<dim3(BB, 2), 256, 0, stream>>>(total, attn_W, tW);
  gemm_qw_kernel<<<1024, 256, 0, stream>>>(x0, Wt, qw);
  attn_kernel<<<LL * BB, 256, 0, stream>>>(f1, f2, f3, f4, f5, feat_emb, qw, tW, x0);

  const size_t pregate_lds = (size_t)(64 * KIN + 64 * HH) * sizeof(_Float16); // 192 KB
  for (int layer = 0; layer < 2; ++layer) {
    (void)hipMemsetAsync(hf16, 0, (size_t)2 * BB * HH * 2, stream);
    (void)hipMemsetAsync(cst,  0, (size_t)2 * BB * HH * 4, stream);
    const _Float16* xin = (layer == 0) ? x0 : y0;
    _Float16* y16 = (layer == 0) ? y0 : nullptr;
    float*    y32 = (layer == 0) ? nullptr : (out + 2 * 131072);
    for (int s = 0; s < LL; ++s) {
      pregate_kernel<<<64, 128, pregate_lds, stream>>>(xin, hf16, wih16, whh16,
                                                       bvec, pre, layer, s);
      lstm_act_kernel<<<256, 256, 0, stream>>>(pre, cst, hf16, y16, y32,
                                               out, out + 131072, layer, s);
    }
  }
}
